// PhonemeAlignmentModel_24489903522589
// MI455X (gfx1250) — matrix operand order repaired and checked by reading, not compiled, not run on hardware
//
#include <hip/hip_runtime.h>
#include <hip/hip_bf16.h>

// ---------------------------------------------------------------------------
// PhonemeAlignmentModel forward for MI455X (gfx1250, wave32, WMMA bf16)
// ---------------------------------------------------------------------------
#define BB 16
#define AD 80
#define LDIM 52
#define HH 512
#define ZZ 128
#define TA 2000
#define TL 200
#define SIGMA 30.0f

typedef __bf16 bf16;
typedef __attribute__((ext_vector_type(16))) __bf16 v16bf;
typedef __attribute__((ext_vector_type(8)))  __bf16 v8bf;   // 16 bytes -> b128 load
typedef __attribute__((ext_vector_type(8)))  float  v8f;

// ---------------------------------------------------------------------------
// Vector fragment loaders (CDNA5 ISA 7.12.2 layouts, wave32).
// A (16x32): lane L<16 -> M=L, elems 0..7 = K 0..7, 8..15 = K 16..23;
//            lane L>=16 -> M=L-16, K 8..15 / 24..31.  With K contiguous in
//            memory this is two 16-byte loads per lane.
// B (32x16): lane%16 = N; lanes 0-15 hold K 0..7/16..23, lanes 16-31 K 8..15/24..31.
//            With K contiguous ([N][K] layout) this is two 16-byte loads.
// C/D (16x16 f32): lane = N, vgpr v -> M = v + 8*(lane/16).
// ---------------------------------------------------------------------------
__device__ __forceinline__ v16bf load_a_vec(const bf16* base, int lane, long rowStride) {
  int m  = lane & 15;
  int kg = (lane >> 4) & 1;
  const bf16* p = base + (long)m * rowStride + (kg << 3);
  v8bf lo = *(const v8bf*)(p);        // K = kg*8 + 0..7
  v8bf hi = *(const v8bf*)(p + 16);   // K = 16 + kg*8 + 0..7
  v16bf a;
#pragma unroll
  for (int e = 0; e < 8; ++e) { a[e] = lo[e]; a[e + 8] = hi[e]; }
  return a;
}

// layout [N][K]: element = base[n*colStride + k], K contiguous
__device__ __forceinline__ v16bf load_b_vec(const bf16* base, int lane,
                                            long colStride, int nmax) {
  int n  = lane & 15;
  int kg = (lane >> 4) & 1;
  v16bf b;
  if (n < nmax) {
    const bf16* p = base + (long)n * colStride + (kg << 3);
    v8bf lo = *(const v8bf*)(p);
    v8bf hi = *(const v8bf*)(p + 16);
#pragma unroll
    for (int e = 0; e < 8; ++e) { b[e] = lo[e]; b[e + 8] = hi[e]; }
  } else {
#pragma unroll
    for (int e = 0; e < 16; ++e) b[e] = (bf16)0.0f;
  }
  return b;
}

// scalar guarded fallbacks (K tails only)
__device__ __forceinline__ v16bf load_a_tail(const bf16* base, int lane,
                                             long rowStride, int kmax) {
  int m  = lane & 15;
  int kg = (lane >> 4) & 1;
  v16bf a;
#pragma unroll
  for (int e = 0; e < 16; ++e) {
    int k = ((e >> 3) << 4) + (kg << 3) + (e & 7);
    bf16 v = (bf16)0.0f;
    if (k < kmax) v = base[(long)m * rowStride + k];
    a[e] = v;
  }
  return a;
}

__device__ __forceinline__ v16bf load_b_tail(const bf16* base, int lane,
                                             long colStride, int kmax, int nmax) {
  int n  = lane & 15;
  int kg = (lane >> 4) & 1;
  v16bf b;
#pragma unroll
  for (int e = 0; e < 16; ++e) {
    int k = ((e >> 3) << 4) + (kg << 3) + (e & 7);
    bf16 v = (bf16)0.0f;
    if (k < kmax && n < nmax) v = base[(long)n * colStride + k];
    b[e] = v;
  }
  return b;
}

__device__ __forceinline__ float wave_reduce_sum(float v) {
#pragma unroll
  for (int off = 16; off; off >>= 1) v += __shfl_down(v, off, 32);
  return v;
}

// ---------------------------------------------------------------------------
// utility kernels
// ---------------------------------------------------------------------------
__global__ void zero_scal_kernel(float* s) {
  if (threadIdx.x < 8) s[threadIdx.x] = 0.0f;
}

__global__ void cvt_bf16_kernel(const float* __restrict__ in, bf16* __restrict__ out, long n) {
  for (long i = (long)blockIdx.x * blockDim.x + threadIdx.x; i < n;
       i += (long)gridDim.x * blockDim.x)
    out[i] = (bf16)in[i];
}

// W [Cout][Cin][3] (f32) -> Wt [3][Cout][Cinp] (bf16, zero padded K)
__global__ void repack_w_kernel(const float* __restrict__ W, bf16* __restrict__ Wt,
                                int Cout, int Cin, int Cinp) {
  long n = 3L * Cout * Cinp;
  for (long i = (long)blockIdx.x * blockDim.x + threadIdx.x; i < n;
       i += (long)gridDim.x * blockDim.x) {
    int k = (int)(i / ((long)Cout * Cinp));
    long rem = i % ((long)Cout * Cinp);
    int m = (int)(rem / Cinp);
    int c = (int)(rem % Cinp);
    bf16 v = (bf16)0.0f;
    if (c < Cin) v = (bf16)W[((long)m * Cin + c) * 3 + k];
    Wt[i] = v;
  }
}

// ---------------------------------------------------------------------------
// conv1d(k=3, SAME) as 3 shifted WMMA GEMMs.  X:[B][Cin][T] bf16,
// Wt:[3][Cout][Cinp] bf16 (padded), bias f32.
// Block = 128 thr = 4 waves; wave tile = 16(Cout) x 32(T), A reused by 2 B's.
// LDS tile time-major [34][40]: row j = t0-1+j, K(channel) contiguous.
// ---------------------------------------------------------------------------
__global__ void conv_wmma_kernel(const bf16* __restrict__ X, const bf16* __restrict__ Wt,
                                 const float* __restrict__ bias,
                                 bf16* Ybf, float* Yf32,
                                 int Cin, int Cinp, int Cout, int T, int relu) {
  const int lane = threadIdx.x & 31;
  const int wave = threadIdx.x >> 5;
  const int b    = blockIdx.z;
  const int t0   = blockIdx.x * 32;
  const int m0   = blockIdx.y * 64 + wave * 16;

  const bf16* Xb = X + (size_t)b * Cin * T;
  __shared__ bf16 Xs[34][40];     // rows: t0-1 .. t0+32 ; cols: 32 channels (pitch 40 = 80B)

  v8f acc0 = {}, acc1 = {};
  const int nK = Cinp >> 5;
  for (int step = 0; step < nK; ++step) {
    const int c0 = step << 5;
    __syncthreads();
    for (int idx = threadIdx.x; idx < 34 * 32; idx += 128) {
      int j = idx % 34, r = idx / 34;          // j: time, r: channel
      int c = c0 + r, t = t0 - 1 + j;
      bf16 v = (bf16)0.0f;
      if (c < Cin && t >= 0 && t < T) v = Xb[(size_t)c * T + t];
      Xs[j][r] = v;
    }
    __syncthreads();
#pragma unroll
    for (int k = 0; k < 3; ++k) {
      const bf16* Wk = Wt + ((size_t)k * Cout + m0) * Cinp + c0;
      if (k == 0 && step + 1 < nK) __builtin_prefetch(Wk + 32, 0, 1);
      v16bf a  = load_a_vec(Wk, lane, Cinp);
      v16bf b0 = load_b_vec(&Xs[k][0],      lane, 40, 16);
      v16bf b1 = load_b_vec(&Xs[k + 16][0], lane, 40, 16);
      acc0 = __builtin_amdgcn_wmma_f32_16x16x32_bf16(false, a, false, b0,
                                                     (short)0, acc0, false, false);
      acc1 = __builtin_amdgcn_wmma_f32_16x16x32_bf16(false, a, false, b1,
                                                     (short)0, acc1, false, false);
    }
  }

  const int n  = lane & 15;
  const int mg = (lane >> 4) << 3;
#pragma unroll
  for (int v = 0; v < 8; ++v) {
    int m = m0 + mg + v;
    float bv = bias[m];
    float val0 = acc0[v] + bv;
    float val1 = acc1[v] + bv;
    if (relu) { val0 = fmaxf(val0, 0.0f); val1 = fmaxf(val1, 0.0f); }
    int ta = t0 + n, tb = t0 + 16 + n;
    size_t rowo = ((size_t)b * Cout + m) * T;
    if (ta < T) { if (Ybf) Ybf[rowo + ta] = (bf16)val0; if (Yf32) Yf32[rowo + ta] = val0; }
    if (tb < T) { if (Ybf) Ybf[rowo + tb] = (bf16)val1; if (Yf32) Yf32[rowo + tb] = val1; }
  }
}

// ---------------------------------------------------------------------------
// KLD inner sum:  accum += sum(1 + lv - mu^2 - exp(lv));  mulv:[B][2Z][T] f32
// ---------------------------------------------------------------------------
__global__ void kld_reduce_kernel(const float* __restrict__ mulv, int T, float* accum) {
  long N = (long)BB * ZZ * T;
  float part = 0.0f;
  for (long i = (long)blockIdx.x * blockDim.x + threadIdx.x; i < N;
       i += (long)gridDim.x * blockDim.x) {
    long t  = i % T;
    long z  = (i / T) % ZZ;
    long bb = i / ((long)T * ZZ);
    float mu = mulv[((long)bb * 2 * ZZ + z) * T + t];
    float lv = mulv[((long)bb * 2 * ZZ + ZZ + z) * T + t];
    part += 1.0f + lv - mu * mu - expf(lv);
  }
  part = wave_reduce_sum(part);
  if ((threadIdx.x & 31) == 0) atomicAdd(accum, part);
}

// mu (first Z chans) -> TRANSPOSED bf16 [B][T][Z] (K=z contiguous) + norms
__global__ void mu_extract_norms_kernel(const float* __restrict__ mulv,
                                        bf16* __restrict__ mut,
                                        float* __restrict__ norms, int T) {
  long i = (long)blockIdx.x * blockDim.x + threadIdx.x;
  if (i >= (long)BB * T) return;
  long bb = i / T, t = i % T;
  float s = 0.0f;
  for (int z = 0; z < ZZ; ++z) {
    float m = mulv[((long)bb * 2 * ZZ + z) * T + t];
    s += m * m;
    mut[((long)bb * T + t) * ZZ + z] = (bf16)m;
  }
  norms[i] = s;
}

// ---------------------------------------------------------------------------
// logits[b,t,s] = -sqrt(max(||a||^2+||l||^2-2<a,l>,0));  muAt:[B][TA][Z],
// muLt:[B][TL][Z] (both K=z contiguous -> pure b128 fragment loads).
// ---------------------------------------------------------------------------
__global__ void cdist_wmma_kernel(const bf16* __restrict__ muAt, const bf16* __restrict__ muLt,
                                  const float* __restrict__ nA, const float* __restrict__ nL,
                                  float* __restrict__ Gout) {
  const int lane  = threadIdx.x & 31;
  const int wave  = threadIdx.x >> 5;
  const int b     = blockIdx.z;
  const int ttile = blockIdx.x * 4 + wave;
  if (ttile >= TA / 16) return;
  const int t0 = ttile * 16;
  const int s0 = blockIdx.y * 16;

  const bf16* Ab = muAt + ((size_t)b * TA + t0) * ZZ;
  const bf16* Lb = muLt + ((size_t)b * TL + s0) * ZZ;

  v8f acc = {};
#pragma unroll
  for (int z0 = 0; z0 < ZZ; z0 += 32) {
    v16bf a  = load_a_vec(Ab + z0, lane, ZZ);
    v16bf bb = load_b_vec(Lb + z0, lane, ZZ, TL - s0);
    acc = __builtin_amdgcn_wmma_f32_16x16x32_bf16(false, a, false, bb,
                                                  (short)0, acc, false, false);
  }

  const int n  = lane & 15;
  const int mg = (lane >> 4) << 3;
  const int s  = s0 + n;
  if (s < TL) {
    float ns = nL[b * TL + s];
#pragma unroll
    for (int v = 0; v < 8; ++v) {
      int t = t0 + mg + v;
      float d2 = nA[b * TA + t] + ns - 2.0f * acc[v];
      Gout[((size_t)b * TA + t) * TL + s] = -sqrtf(fmaxf(d2, 0.0f));
    }
  }
}

// one wave per (b,t) row: logp in place, probs f32 -> d_out,
// probs bf16 TRANSPOSED -> Pbt [B][TL][TA] (K=t contiguous for gamma GEMM)
__global__ void softmax_rows_kernel(float* G, float* probs_f32, bf16* Pbt) {
  const int lane = threadIdx.x & 31;
  const int wave = threadIdx.x >> 5;
  long row = (long)blockIdx.x * 4 + wave;
  if (row >= (long)BB * TA) return;
  float* x = G + row * TL;
  long b = row / TA, t = row % TA;

  float mx = -1e30f;
  for (int s = lane; s < TL; s += 32) mx = fmaxf(mx, x[s]);
#pragma unroll
  for (int off = 16; off; off >>= 1) mx = fmaxf(mx, __shfl_down(mx, off, 32));
  mx = __shfl(mx, 0, 32);

  float sm = 0.0f;
  for (int s = lane; s < TL; s += 32) sm += expf(x[s] - mx);
  sm = wave_reduce_sum(sm);
  sm = __shfl(sm, 0, 32);
  const float lse = mx + logf(sm);

  for (int s = lane; s < TL; s += 32) {
    float lp = x[s] - lse;
    float p  = expf(lp);
    x[s] = lp;                                    // logp in place
    probs_f32[row * TL + s] = p;
    Pbt[((long)b * TL + s) * TA + t] = (bf16)p;   // transposed
  }
}

// gsum = sum_t exp(-t^2 / (2 sigma^2))
__global__ void gsum_kernel(float* out) {
  float s = 0.0f;
  for (int t = threadIdx.x; t < TA; t += 256) {
    float u = (float)t;
    s += expf(-u * u * (1.0f / (2.0f * SIGMA * SIGMA)));
  }
  s = wave_reduce_sum(s);
  __shared__ float red[8];
  if ((threadIdx.x & 31) == 0) red[threadIdx.x >> 5] = s;
  __syncthreads();
  if (threadIdx.x == 0) {
    float tot = 0.0f;
    for (int i = 0; i < 8; ++i) tot += red[i];
    out[0] = tot;
  }
}

// K[s][t] = (0 <= t-s+999 < TA) ? g[t-s+999]/gsum : 0   (bf16)
__global__ void build_K_kernel(const float* __restrict__ gsum, bf16* __restrict__ Kmat) {
  float inv = 1.0f / gsum[0];
  long n = (long)TA * TA;
  for (long i = (long)blockIdx.x * blockDim.x + threadIdx.x; i < n;
       i += (long)gridDim.x * blockDim.x) {
    int s = (int)(i / TA), t = (int)(i % TA);
    int idx = t - s + (TA - 1) / 2;
    float v = 0.0f;
    if (idx >= 0 && idx < TA) {
      float u = (float)idx;
      v = expf(-u * u * (1.0f / (2.0f * SIGMA * SIGMA))) * inv;
    }
    Kmat[i] = (bf16)v;
  }
}

// ---------------------------------------------------------------------------
// gamma = K @ probs (per batch) fused with accum += sum(gamma * logp)
// Kmat:[TA][TA] (K=t contiguous), Pbt:[B][TL][TA] (K=t contiguous).
// 62 vectorized K-steps + 1 guarded scalar tail (t0=1984, 16 left).
// ---------------------------------------------------------------------------
__global__ void gamma_loss_wmma_kernel(const bf16* __restrict__ Kmat,
                                       const bf16* __restrict__ Pbt,
                                       const float* __restrict__ logp, float* accum) {
  const int lane  = threadIdx.x & 31;
  const int wave  = threadIdx.x >> 5;
  const int b     = blockIdx.z;
  const int stile = blockIdx.x * 4 + wave;
  if (stile >= TA / 16) return;
  const int s0 = stile * 16;
  const int o0 = blockIdx.y * 16;
  const bf16* Ks = Kmat + (size_t)s0 * TA;
  const bf16* Pb = Pbt + ((size_t)b * TL + o0) * TA;
  const int nmax = TL - o0;

  v8f acc = {};
  const int kFull = (TA / 32) * 32;   // 1984
  for (int t0 = 0; t0 < kFull; t0 += 32) {
    if (t0 + 32 < kFull) __builtin_prefetch(Ks + t0 + 32, 0, 1);
    v16bf a  = load_a_vec(Ks + t0, lane, TA);
    v16bf bb = load_b_vec(Pb + t0, lane, TA, nmax);
    acc = __builtin_amdgcn_wmma_f32_16x16x32_bf16(false, a, false, bb,
                                                  (short)0, acc, false, false);
  }
  {   // tail: K = 1984..1999 (16 valid)
    v16bf a  = load_a_tail(Ks + kFull, lane, TA, TA - kFull);
    v16bf bb = load_b_tail(Pb + kFull, lane, TA, TA - kFull, nmax);
    acc = __builtin_amdgcn_wmma_f32_16x16x32_bf16(false, a, false, bb,
                                                  (short)0, acc, false, false);
  }

  const int n  = lane & 15;
  const int mg = (lane >> 4) << 3;
  const int o  = o0 + n;
  float part = 0.0f;
  if (o < TL) {
#pragma unroll
    for (int v = 0; v < 8; ++v) {
      int s = s0 + mg + v;
      part += acc[v] * logp[((size_t)b * TA + s) * TL + o];
    }
  }
  part = wave_reduce_sum(part);
  if (lane == 0) atomicAdd(accum, part);
}

// total = -sum(gamma*logp) + 0.1*( -0.5*sumA - 0.5*sumL )
__global__ void finalize_kernel(const float* scal, float* out) {
  if (threadIdx.x == 0 && blockIdx.x == 0)
    out[0] = -scal[2] + 0.1f * (-0.5f * scal[0] - 0.5f * scal[1]);
}

// ---------------------------------------------------------------------------
static inline size_t alignup(size_t x) { return (x + 255) & ~(size_t)255; }

extern "C" void kernel_launch(void* const* d_in, const int* in_sizes, int n_in,
                              void* d_out, int out_size, void* d_ws, size_t ws_size,
                              hipStream_t stream) {
  (void)in_sizes; (void)n_in; (void)out_size; (void)ws_size;
  const float* ain  = (const float*)d_in[0];
  const float* lin  = (const float*)d_in[1];
  const float* a_w1 = (const float*)d_in[2];
  const float* a_b1 = (const float*)d_in[3];
  const float* a_w2 = (const float*)d_in[4];
  const float* a_b2 = (const float*)d_in[5];
  const float* a_w3 = (const float*)d_in[6];
  const float* a_b3 = (const float*)d_in[7];
  const float* l_w1 = (const float*)d_in[8];
  const float* l_b1 = (const float*)d_in[9];
  const float* l_w2 = (const float*)d_in[10];
  const float* l_b2 = (const float*)d_in[11];
  const float* l_w3 = (const float*)d_in[12];
  const float* l_b3 = (const float*)d_in[13];

  const int ADp = 96, LDp = 64;       // Cin padded to 32

  char* w = (char*)d_ws;
  auto take = [&](size_t bytes) -> void* { void* p = (void*)w; w += alignup(bytes); return p; };

  float* scal  = (float*)take(256);   // [0]=sumA [1]=sumL [2]=sum(gamma*logp) [3]=gsum
  bf16* Xa     = (bf16*)take((size_t)BB * AD * TA * 2);
  bf16* Xl     = (bf16*)take((size_t)BB * LDIM * TL * 2);
  bf16* Wa1    = (bf16*)take((size_t)3 * HH * ADp * 2);
  bf16* Wa2    = (bf16*)take((size_t)3 * HH * HH * 2);
  bf16* Wa3    = (bf16*)take((size_t)3 * 2 * ZZ * HH * 2);
  bf16* Wl1    = (bf16*)take((size_t)3 * HH * LDp * 2);
  bf16* Wl2    = (bf16*)take((size_t)3 * HH * HH * 2);
  bf16* Wl3    = (bf16*)take((size_t)3 * 2 * ZZ * HH * 2);
  bf16* h1     = (bf16*)take((size_t)BB * HH * TA * 2);
  bf16* h2     = (bf16*)take((size_t)BB * HH * TA * 2);
  float* mulvA = (float*)take((size_t)BB * 2 * ZZ * TA * 4);
  bf16* hl1    = (bf16*)take((size_t)BB * HH * TL * 2);
  bf16* hl2    = (bf16*)take((size_t)BB * HH * TL * 2);
  float* mulvL = (float*)take((size_t)BB * 2 * ZZ * TL * 4);
  bf16* muAt   = (bf16*)take((size_t)BB * TA * ZZ * 2);   // [b][t][z]
  bf16* muLt   = (bf16*)take((size_t)BB * TL * ZZ * 2);   // [b][s][z]
  float* nA    = (float*)take((size_t)BB * TA * 4);
  float* nL    = (float*)take((size_t)BB * TL * 4);
  float* G     = (float*)take((size_t)BB * TA * TL * 4);  // logits -> logp
  bf16* Kmat   = (bf16*)take((size_t)TA * TA * 2);
  bf16* Pbt    = (bf16*)take((size_t)BB * TL * TA * 2);   // [b][o][t]

  float* out_loss  = (float*)d_out;
  float* out_probs = (float*)d_out + 1;

  zero_scal_kernel<<<1, 32, 0, stream>>>(scal);

  cvt_bf16_kernel<<<2048, 256, 0, stream>>>(ain, Xa, (long)BB * AD * TA);
  cvt_bf16_kernel<<<256, 256, 0, stream>>>(lin, Xl, (long)BB * LDIM * TL);

  repack_w_kernel<<<576, 256, 0, stream>>>(a_w1, Wa1, HH, AD, ADp);
  repack_w_kernel<<<2048, 256, 0, stream>>>(a_w2, Wa2, HH, HH, HH);
  repack_w_kernel<<<1024, 256, 0, stream>>>(a_w3, Wa3, 2 * ZZ, HH, HH);
  repack_w_kernel<<<384, 256, 0, stream>>>(l_w1, Wl1, HH, LDIM, LDp);
  repack_w_kernel<<<2048, 256, 0, stream>>>(l_w2, Wl2, HH, HH, HH);
  repack_w_kernel<<<1024, 256, 0, stream>>>(l_w3, Wl3, 2 * ZZ, HH, HH);

  // acoustic encoder: T tiles of 32 -> 63 ; Cout tiles of 64
  conv_wmma_kernel<<<dim3((TA + 31) / 32, HH / 64, BB), 128, 0, stream>>>(
      Xa, Wa1, a_b1, h1, nullptr, AD, ADp, HH, TA, 1);
  conv_wmma_kernel<<<dim3((TA + 31) / 32, HH / 64, BB), 128, 0, stream>>>(
      h1, Wa2, a_b2, h2, nullptr, HH, HH, HH, TA, 1);
  conv_wmma_kernel<<<dim3((TA + 31) / 32, (2 * ZZ) / 64, BB), 128, 0, stream>>>(
      h2, Wa3, a_b3, nullptr, mulvA, HH, HH, 2 * ZZ, TA, 0);

  // linguistic encoder: 7 T-tiles
  conv_wmma_kernel<<<dim3((TL + 31) / 32, HH / 64, BB), 128, 0, stream>>>(
      Xl, Wl1, l_b1, hl1, nullptr, LDIM, LDp, HH, TL, 1);
  conv_wmma_kernel<<<dim3((TL + 31) / 32, HH / 64, BB), 128, 0, stream>>>(
      hl1, Wl2, l_b2, hl2, nullptr, HH, HH, HH, TL, 1);
  conv_wmma_kernel<<<dim3((TL + 31) / 32, (2 * ZZ) / 64, BB), 128, 0, stream>>>(
      hl2, Wl3, l_b3, nullptr, mulvL, HH, HH, 2 * ZZ, TL, 0);

  kld_reduce_kernel<<<2048, 256, 0, stream>>>(mulvA, TA, scal + 0);
  kld_reduce_kernel<<<512, 256, 0, stream>>>(mulvL, TL, scal + 1);

  mu_extract_norms_kernel<<<(BB * TA + 255) / 256, 256, 0, stream>>>(mulvA, muAt, nA, TA);
  mu_extract_norms_kernel<<<(BB * TL + 255) / 256, 256, 0, stream>>>(mulvL, muLt, nL, TL);

  cdist_wmma_kernel<<<dim3((TA / 16 + 3) / 4, (TL + 15) / 16, BB), 128, 0, stream>>>(
      muAt, muLt, nA, nL, G);

  softmax_rows_kernel<<<(BB * TA) / 4, 128, 0, stream>>>(G, out_probs, Pbt);

  gsum_kernel<<<1, 256, 0, stream>>>(scal + 3);
  build_K_kernel<<<4096, 256, 0, stream>>>(scal + 3, Kmat);

  gamma_loss_wmma_kernel<<<dim3((TA / 16 + 3) / 4, (TL + 15) / 16, BB), 128, 0, stream>>>(
      Kmat, Pbt, G, scal + 2);

  finalize_kernel<<<1, 32, 0, stream>>>(scal, out_loss);
}
